// DeeperGCN_35820027248884
// MI455X (gfx1250) — matrix operand order repaired and checked by reading, not compiled, not run on hardware
//
#include <hip/hip_runtime.h>
#include <hip/hip_bf16.h>

#define NNODES 50000
#define NEDGES 400000
#define NGRAPH 1024
#define CH     128
#define CH2    256
#define NLAYER 7
#define EPS_F  1e-7f

typedef __attribute__((ext_vector_type(16))) __bf16 v16bf;
typedef __attribute__((ext_vector_type(8)))  float  v8f;

union FragU { v16bf bf; uint4 q[2]; };

__device__ __forceinline__ unsigned short f2bf(float f) {
  unsigned int u = __float_as_uint(f);
  u += 0x7fffu + ((u >> 16) & 1u);          // round-to-nearest-even
  return (unsigned short)(u >> 16);
}

__device__ __forceinline__ float bn_apply(float v, float s, float q, float invM,
                                          float g, float b) {
  float mean = s * invM;
  float var  = q * invM - mean * mean;
  return (v - mean) * rsqrtf(var + 1e-5f) * g + b;
}

// ---------------- embeddings ----------------
__global__ void encode_nodes(const int* __restrict__ x,
                             const float* __restrict__ atom_emb,
                             const float* __restrict__ vn_emb,
                             float* __restrict__ h) {
  int n = blockIdx.x, c = threadIdx.x;
  float s = vn_emb[c];
#pragma unroll
  for (int k = 0; k < 9; ++k) {
    int idx = x[n * 9 + k];
    s += atom_emb[(size_t)(k * 128 + idx) * CH + c];
  }
  h[(size_t)n * CH + c] = s;
}

__global__ void init_vn(const float* __restrict__ vn_emb, float* __restrict__ vn) {
  size_t i = (size_t)blockIdx.x * blockDim.x + threadIdx.x;
  vn[i] = vn_emb[i & (CH - 1)];
}

// ---------------- message passing ----------------
__global__ void init_accum(float* __restrict__ zmax, float* __restrict__ den,
                           float* __restrict__ num) {
  size_t i = (size_t)blockIdx.x * blockDim.x + threadIdx.x;
  zmax[i] = -1e30f; den[i] = 0.f; num[i] = 0.f;
}

__global__ void edge_max(const float* __restrict__ hin, const int* __restrict__ ei,
                         const int* __restrict__ ea, const float* __restrict__ bond,
                         float* __restrict__ zmax) {
  int e = blockIdx.x, c = threadIdx.x;
  int src = ei[e], dst = ei[NEDGES + e];
  int a0 = ea[e * 3 + 0], a1 = ea[e * 3 + 1], a2 = ea[e * 3 + 2];
  float eb = bond[(size_t)(0 * 8 + a0) * CH + c] +
             bond[(size_t)(1 * 8 + a1) * CH + c] +
             bond[(size_t)(2 * 8 + a2) * CH + c];
  float m = fmaxf(hin[(size_t)src * CH + c] + eb, 0.f) + EPS_F;
  __hip_atomic_fetch_max(&zmax[(size_t)dst * CH + c], m,
                         __ATOMIC_RELAXED, __HIP_MEMORY_SCOPE_AGENT);
}

__global__ void edge_sum(const float* __restrict__ hin, const int* __restrict__ ei,
                         const int* __restrict__ ea, const float* __restrict__ bond,
                         const float* __restrict__ zmax,
                         float* __restrict__ den, float* __restrict__ num) {
  int e = blockIdx.x, c = threadIdx.x;
  int src = ei[e], dst = ei[NEDGES + e];
  int a0 = ea[e * 3 + 0], a1 = ea[e * 3 + 1], a2 = ea[e * 3 + 2];
  float eb = bond[(size_t)(0 * 8 + a0) * CH + c] +
             bond[(size_t)(1 * 8 + a1) * CH + c] +
             bond[(size_t)(2 * 8 + a2) * CH + c];
  float m = fmaxf(hin[(size_t)src * CH + c] + eb, 0.f) + EPS_F;
  size_t di = (size_t)dst * CH + c;
  float ez = __expf(m - zmax[di]);
  atomicAdd(&den[di], ez);
  atomicAdd(&num[di], m * ez);
}

__global__ void node_update(const float* __restrict__ hin, const float* __restrict__ num,
                            const float* __restrict__ den, unsigned short* __restrict__ hh_bf) {
  size_t i = (size_t)blockIdx.x * blockDim.x + threadIdx.x;
  float agg = num[i] / (den[i] + 1e-16f);
  hh_bf[i] = f2bf(hin[i] + agg);
}

// ---------------- WMMA GEMM: out[M,Nout] = A[M,K](bf16) x BT[Nout,K](bf16) + bias (+res)
__global__ void gemm_bf16_wmma(const unsigned short* __restrict__ A,
                               const unsigned short* __restrict__ BT,
                               const float* __restrict__ bias,
                               const float* __restrict__ res,
                               float* __restrict__ out,
                               int M, int K, int Nout) {
  int wave = threadIdx.x >> 5;
  int lane = threadIdx.x & 31;
  int m0 = blockIdx.x * 16;
  int n0 = (blockIdx.y * 4 + wave) * 16;
  if (m0 >= M || n0 >= Nout) return;
  int hi = lane >> 4;
  int ml = lane & 15;

  const unsigned short* arow = A  + (size_t)(m0 + ml) * K;
  const unsigned short* brow = BT + (size_t)(n0 + ml) * K;

  v8f acc = {};
  for (int k0 = 0; k0 < K; k0 += 32) {
    FragU a, b;
    a.q[0] = *(const uint4*)(arow + k0 + hi * 8);        // K = k0+hi*8+0..7
    a.q[1] = *(const uint4*)(arow + k0 + 16 + hi * 8);   // K = k0+hi*8+16..23
    b.q[0] = *(const uint4*)(brow + k0 + hi * 8);        // K = k0+hi*8+0..7
    b.q[1] = *(const uint4*)(brow + k0 + 16 + hi * 8);   // K = k0+hi*8+16..23
    acc = __builtin_amdgcn_wmma_f32_16x16x32_bf16(false, a.bf, false, b.bf,
                                                  (short)0, acc, false, false);
  }

  int n = n0 + ml;
  float bv = bias ? bias[n] : 0.f;
#pragma unroll
  for (int r = 0; r < 8; ++r) {
    int m = m0 + r + hi * 8;
    float v = acc[r] + bv;
    if (res) v += res[(size_t)m * Nout + n];
    out[(size_t)m * Nout + n] = v;
  }
}

// ---------------- batchnorm helpers ----------------
__global__ void colstats(const float* __restrict__ in, float* __restrict__ stats,
                         int M, int Ncol, int rowsPerBlock) {
  int c = threadIdx.x;                       // blockDim.x == Ncol
  int r0 = blockIdx.x * rowsPerBlock;
  int r1 = r0 + rowsPerBlock; if (r1 > M) r1 = M;
  float s = 0.f, q = 0.f;
  for (int r = r0; r < r1; ++r) {
    float v = in[(size_t)r * Ncol + c];
    s += v; q += v * v;
  }
  atomicAdd(&stats[c], s);
  atomicAdd(&stats[Ncol + c], q);
}

__global__ void bn_relu_bf(const float* __restrict__ in, const float* __restrict__ stats,
                           const float* __restrict__ g, const float* __restrict__ b,
                           unsigned short* __restrict__ out, float invM, int cmask) {
  size_t i = (size_t)blockIdx.x * blockDim.x + threadIdx.x;
  int c = (int)(i & cmask);
  float v = bn_apply(in[i], stats[c], stats[cmask + 1 + c], invM, g[c], b[c]);
  out[i] = f2bf(fmaxf(v, 0.f));
}

__global__ void bn_relu_f32(const float* __restrict__ in, const float* __restrict__ stats,
                            const float* __restrict__ g, const float* __restrict__ b,
                            float* __restrict__ out, float invM, int cmask) {
  size_t i = (size_t)blockIdx.x * blockDim.x + threadIdx.x;
  int c = (int)(i & cmask);
  float v = bn_apply(in[i], stats[c], stats[cmask + 1 + c], invM, g[c], b[c]);
  out[i] = fmaxf(v, 0.f);
}

// h2 = relu(bn(h)); also vt[batch] += h2 (vt pre-initialized to vn)
__global__ void bn_relu_pool(const float* __restrict__ hcur, const float* __restrict__ stats,
                             const float* __restrict__ g, const float* __restrict__ b,
                             const int* __restrict__ batch, float* __restrict__ hin,
                             float* __restrict__ vt, float invM) {
  size_t i = (size_t)blockIdx.x * blockDim.x + threadIdx.x;
  int c = (int)(i & (CH - 1));
  int n = (int)(i >> 7);
  float v = fmaxf(bn_apply(hcur[i], stats[c], stats[CH + c], invM, g[c], b[c]), 0.f);
  hin[i] = v;
  atomicAdd(&vt[(size_t)batch[n] * CH + c], v);
}

__global__ void add_vn(float* __restrict__ hin, const float* __restrict__ vn,
                       const int* __restrict__ batch) {
  size_t i = (size_t)blockIdx.x * blockDim.x + threadIdx.x;
  int c = (int)(i & (CH - 1));
  int n = (int)(i >> 7);
  hin[i] += vn[(size_t)batch[n] * CH + c];
}

__global__ void copy_f32(const float* __restrict__ in, float* __restrict__ out) {
  size_t i = (size_t)blockIdx.x * blockDim.x + threadIdx.x;
  out[i] = in[i];
}

__global__ void cvt_bf(const float* __restrict__ in, unsigned short* __restrict__ out) {
  size_t i = (size_t)blockIdx.x * blockDim.x + threadIdx.x;
  out[i] = f2bf(in[i]);
}

__global__ void final_pool(const float* __restrict__ hcur, const float* __restrict__ stats,
                           const float* __restrict__ g, const float* __restrict__ b,
                           const int* __restrict__ batch, float* __restrict__ out, float invM) {
  size_t i = (size_t)blockIdx.x * blockDim.x + threadIdx.x;
  int c = (int)(i & (CH - 1));
  int n = (int)(i >> 7);
  float v = bn_apply(hcur[i], stats[c], stats[CH + c], invM, g[c], b[c]);
  atomicAdd(&out[(size_t)batch[n] * CH + c], v);
}

// weight fp32 [K,Nout] -> bf16 transposed [Nout,K]
__global__ void wconvert(const float* __restrict__ W, unsigned short* __restrict__ BT,
                         int K, int Nout) {
  int n = blockIdx.x;
  for (int k = threadIdx.x; k < K; k += blockDim.x)
    BT[(size_t)n * K + k] = f2bf(W[(size_t)k * Nout + n]);
}

// ---------------- host ----------------
extern "C" void kernel_launch(void* const* d_in, const int* in_sizes, int n_in,
                              void* d_out, int out_size, void* d_ws, size_t ws_size,
                              hipStream_t stream) {
  const int*   x          = (const int*)d_in[0];
  const int*   edge_attr  = (const int*)d_in[1];
  const int*   edge_index = (const int*)d_in[2];
  const int*   batch      = (const int*)d_in[3];
  const float* atom_emb   = (const float*)d_in[4];
  const float* bond_emb   = (const float*)d_in[5];
  const float* vn_emb     = (const float*)d_in[6];
  const float* convW1     = (const float*)d_in[7];
  const float* convb1     = (const float*)d_in[8];
  const float* convg1     = (const float*)d_in[9];
  const float* convbb1    = (const float*)d_in[10];
  const float* convW2     = (const float*)d_in[11];
  const float* convb2     = (const float*)d_in[12];
  const float* normg      = (const float*)d_in[13];
  const float* normb      = (const float*)d_in[14];
  const float* vnW1       = (const float*)d_in[15];
  const float* vnb1       = (const float*)d_in[16];
  const float* vng1       = (const float*)d_in[17];
  const float* vnbb1      = (const float*)d_in[18];
  const float* vnW2       = (const float*)d_in[19];
  const float* vnb2       = (const float*)d_in[20];
  const float* vng2       = (const float*)d_in[21];
  const float* vnbb2      = (const float*)d_in[22];
  float* out = (float*)d_out;

  const size_t NC  = (size_t)NNODES * CH;
  const size_t NC2 = (size_t)NNODES * CH2;
  const size_t GC  = (size_t)NGRAPH * CH;

  uintptr_t p = (uintptr_t)d_ws;
  auto alloc = [&](size_t bytes) -> void* {
    uintptr_t r = (p + 255) & ~(uintptr_t)255;
    p = r + bytes;
    return (void*)r;
  };
  float* h0    = (float*)alloc(NC * 4);
  float* h1    = (float*)alloc(NC * 4);
  float* hin   = (float*)alloc(NC * 4);
  float* num   = (float*)alloc(NC * 4);
  float* den   = (float*)alloc(NC * 4);
  float* zmax  = (float*)alloc(NC * 4);
  float* y1    = (float*)alloc(NC2 * 4);
  unsigned short* hh_bf = (unsigned short*)alloc(NC * 2);
  unsigned short* y1_bf = (unsigned short*)alloc(NC2 * 2);
  float* vn    = (float*)alloc(GC * 4);
  float* vt    = (float*)alloc(GC * 4);
  float* vtmp  = (float*)alloc(GC * 4);
  unsigned short* vt_bf   = (unsigned short*)alloc(GC * 2);
  unsigned short* vtmp_bf = (unsigned short*)alloc(GC * 2);
  float* stats = (float*)alloc(512 * 4);
  unsigned short* W1bf   = (unsigned short*)alloc((size_t)NLAYER * CH * CH2 * 2);
  unsigned short* W2bf   = (unsigned short*)alloc((size_t)NLAYER * CH * CH2 * 2);
  unsigned short* vnW1bf = (unsigned short*)alloc((size_t)(NLAYER - 1) * CH * CH * 2);
  unsigned short* vnW2bf = (unsigned short*)alloc((size_t)(NLAYER - 1) * CH * CH * 2);

  // ---- weight conversion (bf16, transposed) ----
  for (int l = 0; l < NLAYER; ++l) {
    wconvert<<<CH2, 128, 0, stream>>>(convW1 + (size_t)l * CH * CH2,
                                      W1bf + (size_t)l * CH2 * CH, CH, CH2);
    wconvert<<<CH, 128, 0, stream>>>(convW2 + (size_t)l * CH2 * CH,
                                     W2bf + (size_t)l * CH * CH2, CH2, CH);
  }
  for (int l = 0; l < NLAYER - 1; ++l) {
    wconvert<<<CH, 128, 0, stream>>>(vnW1 + (size_t)l * CH * CH,
                                     vnW1bf + (size_t)l * CH * CH, CH, CH);
    wconvert<<<CH, 128, 0, stream>>>(vnW2 + (size_t)l * CH * CH,
                                     vnW2bf + (size_t)l * CH * CH, CH, CH);
  }

  const float invN = 1.f / (float)NNODES;
  const float invG = 1.f / (float)NGRAPH;

  auto run_gen = [&](int l, const float* hin_p, const float* res_p, float* outp) {
    init_accum<<<(int)(NC / 256), 256, 0, stream>>>(zmax, den, num);
    edge_max<<<NEDGES, CH, 0, stream>>>(hin_p, edge_index, edge_attr, bond_emb, zmax);
    edge_sum<<<NEDGES, CH, 0, stream>>>(hin_p, edge_index, edge_attr, bond_emb,
                                        zmax, den, num);
    node_update<<<(int)(NC / 256), 256, 0, stream>>>(hin_p, num, den, hh_bf);
    // GEMM1: [N,128] x [128,256]
    gemm_bf16_wmma<<<dim3(NNODES / 16, CH2 / 64), 128, 0, stream>>>(
        hh_bf, W1bf + (size_t)l * CH2 * CH, convb1 + (size_t)l * CH2,
        nullptr, y1, NNODES, CH, CH2);
    hipMemsetAsync(stats, 0, 512 * 4, stream);
    colstats<<<(NNODES + 63) / 64, CH2, 0, stream>>>(y1, stats, NNODES, CH2, 64);
    bn_relu_bf<<<(int)(NC2 / 256), 256, 0, stream>>>(
        y1, stats, convg1 + (size_t)l * CH2, convbb1 + (size_t)l * CH2,
        y1_bf, invN, CH2 - 1);
    // GEMM2: [N,256] x [256,128] (+ residual)
    gemm_bf16_wmma<<<dim3(NNODES / 16, CH / 64), 128, 0, stream>>>(
        y1_bf, W2bf + (size_t)l * CH * CH2, convb2 + (size_t)l * CH,
        res_p, outp, NNODES, CH2, CH);
  };

  // ---- encode + initial vn ----
  encode_nodes<<<NNODES, CH, 0, stream>>>(x, atom_emb, vn_emb, hin);
  init_vn<<<(int)(GC / 256), 256, 0, stream>>>(vn_emb, vn);

  // ---- layer 0 ----
  run_gen(0, hin, nullptr, h0);
  float* hCur = h0;
  float* hNext = h1;

  // ---- layers 1..6 ----
  for (int l = 1; l < NLAYER; ++l) {
    // h2 = relu(bn(h)); vt = vn + segment_sum(h2)
    hipMemsetAsync(stats, 0, 512 * 4, stream);
    colstats<<<(NNODES + 63) / 64, CH, 0, stream>>>(hCur, stats, NNODES, CH, 64);
    copy_f32<<<(int)(GC / 256), 256, 0, stream>>>(vn, vt);
    bn_relu_pool<<<(int)(NC / 256), 256, 0, stream>>>(
        hCur, stats, normg + (size_t)(l - 1) * CH, normb + (size_t)(l - 1) * CH,
        batch, hin, vt, invN);
    // vn MLP: y = relu(bn(vt@W1+b1)); vn = relu(bn(y@W2+b2))
    cvt_bf<<<(int)(GC / 256), 256, 0, stream>>>(vt, vt_bf);
    gemm_bf16_wmma<<<dim3(NGRAPH / 16, CH / 64), 128, 0, stream>>>(
        vt_bf, vnW1bf + (size_t)(l - 1) * CH * CH, vnb1 + (size_t)(l - 1) * CH,
        nullptr, vtmp, NGRAPH, CH, CH);
    hipMemsetAsync(stats, 0, 512 * 4, stream);
    colstats<<<(NGRAPH + 63) / 64, CH, 0, stream>>>(vtmp, stats, NGRAPH, CH, 64);
    bn_relu_bf<<<(int)(GC / 256), 256, 0, stream>>>(
        vtmp, stats, vng1 + (size_t)(l - 1) * CH, vnbb1 + (size_t)(l - 1) * CH,
        vtmp_bf, invG, CH - 1);
    gemm_bf16_wmma<<<dim3(NGRAPH / 16, CH / 64), 128, 0, stream>>>(
        vtmp_bf, vnW2bf + (size_t)(l - 1) * CH * CH, vnb2 + (size_t)(l - 1) * CH,
        nullptr, vt, NGRAPH, CH, CH);
    hipMemsetAsync(stats, 0, 512 * 4, stream);
    colstats<<<(NGRAPH + 63) / 64, CH, 0, stream>>>(vt, stats, NGRAPH, CH, 64);
    bn_relu_f32<<<(int)(GC / 256), 256, 0, stream>>>(
        vt, stats, vng2 + (size_t)(l - 1) * CH, vnbb2 + (size_t)(l - 1) * CH,
        vn, invG, CH - 1);
    // h2 += vn[batch]; h = gen(l, h2) + h
    add_vn<<<(int)(NC / 256), 256, 0, stream>>>(hin, vn, batch);
    run_gen(l, hin, hCur, hNext);
    float* t = hCur; hCur = hNext; hNext = t;
  }

  // ---- final BN + global_add_pool ----
  hipMemsetAsync(stats, 0, 512 * 4, stream);
  colstats<<<(NNODES + 63) / 64, CH, 0, stream>>>(hCur, stats, NNODES, CH, 64);
  hipMemsetAsync(out, 0, GC * 4, stream);
  final_pool<<<(int)(NC / 256), 256, 0, stream>>>(
      hCur, stats, normg + (size_t)(NLAYER - 1) * CH, normb + (size_t)(NLAYER - 1) * CH,
      batch, out, invN);
}
